// TransformerLayer_15779709846088
// MI455X (gfx1250) — matrix operand order repaired and checked by reading, not compiled, not run on hardware
//
#include <hip/hip_runtime.h>
#include <hip/hip_bf16.h>

typedef __bf16 bf16_t;
typedef __bf16 v8bf  __attribute__((ext_vector_type(8)));
typedef __bf16 v16bf __attribute__((ext_vector_type(16)));
typedef float  v8f   __attribute__((ext_vector_type(8)));

#define RTOT 131072   // B*H*W = 8*128*128
#define CDIM 256

__device__ __forceinline__ v8f vzero8() {
  v8f z = {0.f,0.f,0.f,0.f,0.f,0.f,0.f,0.f};
  return z;
}

__device__ __forceinline__ v8f wmma_bf16(v16bf a, v16bf b, v8f c) {
  // D = A(16x32 bf16) * B(32x16 bf16) + C(16x16 f32)
  return __builtin_amdgcn_wmma_f32_16x16x32_bf16(false, a, false, b, (short)0, c,
                                                 false, false);
}

// CDNA5 async copy: 16 bytes global -> LDS, tracked by ASYNCcnt (no VGPR round trip).
// Low 32 bits of a generic pointer to __shared__ are the wave-relative LDS offset.
__device__ __forceinline__ void async_cp16(void* lds, const void* g) {
  unsigned l = (unsigned)(unsigned long long)lds;
  asm volatile("global_load_async_to_lds_b128 %0, %1, off"
               :: "v"(l), "v"(g) : "memory");
}
__device__ __forceinline__ void async_wait0() {
  asm volatile("s_wait_asynccnt 0x0" ::: "memory");
}

// A fragment (16x32, M-major rows) from row-major [row][k] storage, stride in elems.
// Lane h=lane>>4, r=lane&15: elems 0..7 -> k = 8h+0..7 ; elems 8..15 -> k = 16+8h+0..7
__device__ __forceinline__ v16bf frag_a(const bf16_t* p, int stride) {
  int lane = threadIdx.x & 31;
  int h = lane >> 4, r = lane & 15;
  const bf16_t* q = p + r * stride + h * 8;
  v8bf lo = *(const v8bf*)(q);
  v8bf hi = *(const v8bf*)(q + 16);
  v16bf o;
#pragma unroll
  for (int i = 0; i < 8; i++) { o[i] = lo[i]; o[i + 8] = hi[i]; }
  return o;
}

// B fragment (32x16 KxN) read from N-major [n][k] storage, stride in elems.
// Lane h,r: n = r, elems 0..7 -> k = 8h+0..7 ; elems 8..15 -> k = 16+8h+0..7
__device__ __forceinline__ v16bf frag_b(const bf16_t* p, int stride) {
  int lane = threadIdx.x & 31;
  int h = lane >> 4, r = lane & 15;
  const bf16_t* q = p + r * stride + h * 8;
  v8bf lo = *(const v8bf*)(q);
  v8bf hi = *(const v8bf*)(q + 16);
  v16bf o;
#pragma unroll
  for (int i = 0; i < 8; i++) { o[i] = lo[i]; o[i + 8] = hi[i]; }
  return o;
}

// roll(-16,-16) + window-split row permutation (store side of QKV projections)
__device__ __forceinline__ int qkv_perm_row(int row) {
  int b = row >> 14;
  int rem = row & 16383;
  int hh = rem >> 7, ww = rem & 127;
  int hs = (hh + 112) & 127;          // h - 16 mod 128
  int ws = (ww + 112) & 127;
  int win = (b << 4) + ((hs >> 5) << 2) + (ws >> 5);
  return (win << 10) + ((hs & 31) << 5) + (ws & 31);
}

#define EPI_F32  0
#define EPI_QKV  1
#define EPI_VT   2
#define EPI_GELU 3

// C = A(MxK bf16, lda) * Bt(NxK bf16, pre-transposed weights), fp32 accum via WMMA.
// Block tile 128x128, 8 waves, wave tile 32x64 (2x4 wmma tiles), K step 32.
// Double-buffered LDS, filled exclusively with global_load_async_to_lds_b128.
template <int EPI>
__global__ __launch_bounds__(256)
void gemm_bf16_kernel(const bf16_t* __restrict__ A, int lda,
                      const bf16_t* __restrict__ Bt,
                      void* __restrict__ D, int ldd,
                      int K, float scale) {
  __shared__ bf16_t Ash[2][128 * 32];
  __shared__ bf16_t Bts[2][128 * 32];   // [n][k]
  const int tid = threadIdx.x;
  const int bm = blockIdx.y << 7, bn = blockIdx.x << 7;
  const int wave = tid >> 5, lane = tid & 31;
  const int h = lane >> 4, r = lane & 15;
  const int wr = (wave >> 1) << 5;   // 0,32,64,96
  const int wc = (wave & 1) << 6;    // 0,64

  // staging: each thread owns 16 elems (2 x 16B) of A and of Bt per tile
  const int trow = tid >> 1;             // 0..127
  const int tseg = (tid & 1) << 4;       // 0 or 16 elems
  const bf16_t* Ag = A + (size_t)(bm + trow) * lda + tseg;
  const bf16_t* Bg = Bt + (size_t)(bn + trow) * K + tseg;
  bf16_t* Al = &Ash[0][0] + trow * 32 + tseg;
  bf16_t* Bl = &Bts[0][0] + trow * 32 + tseg;

  v8f acc[2][4];
#pragma unroll
  for (int i = 0; i < 2; i++)
#pragma unroll
    for (int j = 0; j < 4; j++) acc[i][j] = vzero8();

  const int NT = K >> 5;
  // prologue: stage tile 0 into buffer 0
  async_cp16(Al, Ag);
  async_cp16(Al + 8, Ag + 8);
  async_cp16(Bl, Bg);
  async_cp16(Bl + 8, Bg + 8);
  async_wait0();
  __syncthreads();

  for (int t = 0; t < NT; ++t) {
    const int cur = t & 1;
    if (t + 1 < NT) {  // async prefetch of next tile overlaps this tile's WMMAs
      const int nxt = cur ^ 1;
      const int ko = (t + 1) << 5;
      async_cp16(Al + nxt * 4096, Ag + ko);
      async_cp16(Al + nxt * 4096 + 8, Ag + ko + 8);
      async_cp16(Bl + nxt * 4096, Bg + ko);
      async_cp16(Bl + nxt * 4096 + 8, Bg + ko + 8);
    }
    const bf16_t* Ab = &Ash[cur][0];
    const bf16_t* Bb = &Bts[cur][0];
    v16bf af[2];
    af[0] = frag_a(Ab + wr * 32, 32);
    af[1] = frag_a(Ab + (wr + 16) * 32, 32);
    v16bf bfm[4];
#pragma unroll
    for (int j = 0; j < 4; j++) bfm[j] = frag_b(Bb + (wc + j * 16) * 32, 32);
#pragma unroll
    for (int i = 0; i < 2; i++)
#pragma unroll
      for (int j = 0; j < 4; j++) acc[i][j] = wmma_bf16(af[i], bfm[j], acc[i][j]);
    async_wait0();
    __syncthreads();
  }

#pragma unroll
  for (int i = 0; i < 2; i++) {
#pragma unroll
    for (int j = 0; j < 4; j++) {
#pragma unroll
      for (int e = 0; e < 8; e++) {
        int row = bm + wr + i * 16 + e + (h << 3);
        int col = bn + wc + j * 16 + r;
        float v = acc[i][j][e];
        if (EPI == EPI_F32) {
          ((float*)D)[(size_t)row * ldd + col] = v;
        } else if (EPI == EPI_QKV) {
          int prow = qkv_perm_row(row);
          ((bf16_t*)D)[(size_t)prow * CDIM + col] = (bf16_t)(v * scale);
        } else if (EPI == EPI_VT) {
          // V stored per-window transposed: [win][c][key]
          int prow = qkv_perm_row(row);
          int wwin = prow >> 10, idx = prow & 1023;
          ((bf16_t*)D)[(((size_t)wwin << 8) + col << 10) + idx] = (bf16_t)v;
        } else {  // EPI_GELU, exact
          float g = 0.5f * v * (1.0f + erff(v * 0.70710678118654752f));
          ((bf16_t*)D)[(size_t)row * ldd + col] = (bf16_t)g;
        }
      }
    }
  }
}

// Flash window attention: grid (128 windows, 8 q-tiles), 256 threads (8 waves).
// Each wave owns 16 query rows; computes S^T = K*Q^T so probs land directly in
// A-fragment layout for P*V. Streaming softmax with mask. K/V tiles double-
// buffered in LDS via async copies (V pre-transposed per window by EPI_VT).
__global__ __launch_bounds__(256)
void window_attention_kernel(const bf16_t* __restrict__ qw,
                             const bf16_t* __restrict__ kw,
                             const bf16_t* __restrict__ vwt,
                             const float* __restrict__ mask,
                             bf16_t* __restrict__ amsg) {
  __shared__ bf16_t Ks[2][32 * 256];   // [key][c]
  __shared__ bf16_t Vs[2][256 * 32];   // [c][key]
  const int win = blockIdx.x;
  const int q0 = blockIdx.y << 7;
  const int tid = threadIdx.x, wave = tid >> 5, lane = tid & 31;
  const int h = lane >> 4, r = lane & 15;

  const bf16_t* kb = kw + (size_t)win * 1024 * 256;
  const bf16_t* vb = vwt + (size_t)win * 256 * 1024;

  // preload this wave's Q fragments (B-side of S^T GEMM) into registers
  v16bf qf[8];
  const bf16_t* qbase = qw + ((size_t)win * 1024 + q0 + (wave << 4)) * 256;
#pragma unroll
  for (int c = 0; c < 8; c++) qf[c] = frag_b(qbase + c * 32, 256);

  v8f oacc[16];
#pragma unroll
  for (int t = 0; t < 16; t++) oacc[t] = vzero8();
  float m_run = -3.0e38f, l_run = 0.f;

  // this lane's query row for mask/stat purposes is q = lane&15
  const float* mrowbase = mask + ((size_t)(win & 15) << 20) +
                          ((size_t)(q0 + (wave << 4) + r) << 10);

  // prologue: stage key-chunk 0 into buffer 0
#pragma unroll
  for (int it = 0; it < 4; ++it) {
    int idx = (it * 256 + tid) << 3;
    async_cp16(&Ks[0][idx], kb + idx);
  }
#pragma unroll
  for (int it = 0; it < 4; ++it) {
    int e = it * 256 + tid;
    int c = e >> 2, sg = (e & 3) << 3;
    async_cp16(&Vs[0][c * 32 + sg], vb + (size_t)c * 1024 + sg);
  }
  async_wait0();
  __syncthreads();

  for (int kc = 0; kc < 32; ++kc) {
    const int cur = kc & 1;
    if (kc + 1 < 32) {  // async prefetch of next 32-key chunk
      const int nxt = cur ^ 1;
      const size_t ko = (size_t)(kc + 1) * 32;
#pragma unroll
      for (int it = 0; it < 4; ++it) {
        int idx = (it * 256 + tid) << 3;
        async_cp16(&Ks[nxt][idx], kb + ko * 256 + idx);
      }
#pragma unroll
      for (int it = 0; it < 4; ++it) {
        int e = it * 256 + tid;
        int c = e >> 2, sg = (e & 3) << 3;
        async_cp16(&Vs[nxt][c * 32 + sg], vb + (size_t)c * 1024 + ko + sg);
      }
    }

    // S^T tiles: rows = keys (two 16-key tiles), cols = this wave's 16 queries
    v8f s0 = vzero8(), s1 = vzero8();
#pragma unroll
    for (int cc = 0; cc < 8; ++cc) {
      v16bf ak0 = frag_a(&Ks[cur][0] + cc * 32, 256);
      v16bf ak1 = frag_a(&Ks[cur][0] + 16 * 256 + cc * 32, 256);
      s0 = wmma_bf16(ak0, qf[cc], s0);
      s1 = wmma_bf16(ak1, qf[cc], s1);
    }
    // add mask: value (key,q) sits at lane = q + 16*(key>=8), reg = key&7
    const float* mr = mrowbase + kc * 32 + (h << 3);
#pragma unroll
    for (int j = 0; j < 8; j++) {
      s0[j] += mr[j];
      s1[j] += mr[16 + j];
    }
    // streaming softmax for q = lane&15 (stats replicated across the lane pair)
    float mx = fmaxf(s0[0], s1[0]);
#pragma unroll
    for (int j = 1; j < 8; j++) mx = fmaxf(mx, fmaxf(s0[j], s1[j]));
    mx = fmaxf(mx, __shfl_xor(mx, 16, 32));
    float m_new = fmaxf(m_run, mx);
    float alpha = __expf(m_run - m_new);
    v16bf pa;
    float rs = 0.f;
#pragma unroll
    for (int j = 0; j < 8; j++) {
      float p0 = __expf(s0[j] - m_new);
      float p1 = __expf(s1[j] - m_new);
      rs += p0 + p1;
      pa[j] = (bf16_t)p0;      // A-frag elems 0..7  : keys 8h + 0..7
      pa[j + 8] = (bf16_t)p1;  // A-frag elems 8..15 : keys 16 + 8h + 0..7
    }
    rs += __shfl_xor(rs, 16, 32);
    l_run = l_run * alpha + rs;
    m_run = m_new;
    // rescale output accumulators (row of reg j is q = j + 8h)
#pragma unroll
    for (int j = 0; j < 8; j++) {
      float aj = __shfl(alpha, j + (h << 3), 32);
#pragma unroll
      for (int ct = 0; ct < 16; ++ct) oacc[ct][j] *= aj;
    }
    // O += P * V  (K dim = 32 keys, 16 column tiles of 16)
#pragma unroll
    for (int ct = 0; ct < 16; ++ct) {
      v16bf bv = frag_b(&Vs[cur][0] + (ct << 4) * 32, 32);
      oacc[ct] = wmma_bf16(pa, bv, oacc[ct]);
    }
    async_wait0();
    __syncthreads();
  }

  // finalize: divide by l, store with window-merge + roll-back permutation
  const int b = win >> 4;
  const int wy = (win >> 2) & 3;
  const int wx = win & 3;
#pragma unroll
  for (int j = 0; j < 8; j++) {
    float lj = __shfl(l_run, j + (h << 3), 32);
    float inv = 1.0f / lj;
    int qrow = q0 + (wave << 4) + j + (h << 3);
    int iy = qrow >> 5, ix = qrow & 31;
    int hh = ((wy << 5) + iy + 16) & 127;
    int ww = ((wx << 5) + ix + 16) & 127;
    size_t orow = ((size_t)b << 14) + (hh << 7) + ww;
#pragma unroll
    for (int ct = 0; ct < 16; ++ct) {
      int col = (ct << 4) + r;
      amsg[orow * 256 + col] = (bf16_t)(oacc[ct][j] * inv);
    }
  }
}

// One wave per row (C=256). MODE 0: bf16 into concat[:,256:512].
// MODE 1: out = src + LN(raw), fp32.
template <int MODE>
__global__ __launch_bounds__(256)
void layernorm_kernel(const float* __restrict__ raw,
                      const float* __restrict__ gamma,
                      const float* __restrict__ beta,
                      const float* __restrict__ src,
                      void* __restrict__ out) {
  const int wave = threadIdx.x >> 5, lane = threadIdx.x & 31;
  const size_t row = ((size_t)blockIdx.x << 3) + wave;
  const float* p = raw + row * 256;
  float4 x0 = *(const float4*)(p + (lane << 2));
  float4 x1 = *(const float4*)(p + 128 + (lane << 2));
  float xv[8] = {x0.x, x0.y, x0.z, x0.w, x1.x, x1.y, x1.z, x1.w};
  float s = 0.f, s2 = 0.f;
#pragma unroll
  for (int k = 0; k < 8; k++) { s += xv[k]; s2 += xv[k] * xv[k]; }
#pragma unroll
  for (int off = 16; off >= 1; off >>= 1) {
    s += __shfl_xor(s, off, 32);
    s2 += __shfl_xor(s2, off, 32);
  }
  float mean = s * (1.0f / 256.0f);
  float var = s2 * (1.0f / 256.0f) - mean * mean;
  float rstd = rsqrtf(var + 1e-5f);
#pragma unroll
  for (int k = 0; k < 8; k++) {
    int c = (k < 4) ? ((lane << 2) + k) : (128 + (lane << 2) + k - 4);
    float y = (xv[k] - mean) * rstd * gamma[c] + beta[c];
    if (MODE == 0) {
      ((bf16_t*)out)[row * 512 + 256 + c] = (bf16_t)y;
    } else {
      ((float*)out)[row * 256 + c] = src[row * 256 + c] + y;
    }
  }
}

__global__ void cvt_f32_bf16_kernel(const float* __restrict__ in,
                                    bf16_t* __restrict__ out, int n) {
  int i = blockIdx.x * 256 + threadIdx.x;
  if (i < n) out[i] = (bf16_t)in[i];
}

__global__ void cvt_f32_bf16_strided_kernel(const float* __restrict__ in,
                                            bf16_t* __restrict__ out,
                                            int cols, int ldo, int n) {
  int i = blockIdx.x * 256 + threadIdx.x;
  if (i < n) {
    int rr = i / cols, cc = i - rr * cols;
    out[(size_t)rr * ldo + cc] = (bf16_t)in[i];
  }
}

// fp32 [K][N] -> bf16 [N][K] (weights pre-transposed so GEMM B staging is a
// straight async copy, no ds_store scatter)
__global__ void cvt_transpose_bf16_kernel(const float* __restrict__ in,
                                          bf16_t* __restrict__ out,
                                          int Kd, int Nd) {
  int i = blockIdx.x * 256 + threadIdx.x;
  if (i < Kd * Nd) {
    int k = i / Nd, n = i - k * Nd;
    out[(size_t)n * Kd + k] = (bf16_t)in[i];
  }
}

extern "C" void kernel_launch(void* const* d_in, const int* in_sizes, int n_in,
                              void* d_out, int out_size, void* d_ws, size_t ws_size,
                              hipStream_t stream) {
  (void)in_sizes; (void)n_in; (void)out_size; (void)ws_size;
  const float* src = (const float*)d_in[0];
  const float* tgt = (const float*)d_in[1];
  const float* mask = (const float*)d_in[2];
  const float* wq = (const float*)d_in[3];
  const float* wk = (const float*)d_in[4];
  const float* wv = (const float*)d_in[5];
  const float* wm = (const float*)d_in[6];
  const float* g1 = (const float*)d_in[7];
  const float* b1 = (const float*)d_in[8];
  const float* w1 = (const float*)d_in[9];
  const float* w2 = (const float*)d_in[10];
  const float* g2 = (const float*)d_in[11];
  const float* b2 = (const float*)d_in[12];
  float* out = (float*)d_out;

  char* ws = (char*)d_ws;
  size_t off = 0;
  auto alloc = [&](size_t bytes) -> char* {
    char* p = ws + off;
    off += (bytes + 255) & ~(size_t)255;
    return p;
  };
  const size_t R = RTOT;
  bf16_t* concat = (bf16_t*)alloc(R * 512 * 2);          // [src_bf16 | msg_bf16]
  bf16_t* tgtb = (bf16_t*)alloc(R * 256 * 2);
  bf16_t* qwb = (bf16_t*)alloc(R * 256 * 2);             // windowed
  bf16_t* kwb = (bf16_t*)alloc(R * 256 * 2);
  bf16_t* vwt = (bf16_t*)alloc(R * 256 * 2);             // windowed, transposed
  bf16_t* amsg = (bf16_t*)alloc(R * 256 * 2);            // un-windowed attn out
  float* mraw = (float*)alloc(R * 256 * 4);              // reused LN input
  bf16_t* hid = (bf16_t*)alloc((size_t)16384 * 2048 * 2);
  bf16_t* wqt = (bf16_t*)alloc(256 * 256 * 2);
  bf16_t* wkt = (bf16_t*)alloc(256 * 256 * 2);
  bf16_t* wvt = (bf16_t*)alloc(256 * 256 * 2);
  bf16_t* wmt = (bf16_t*)alloc(256 * 256 * 2);
  bf16_t* w1t = (bf16_t*)alloc(512 * 2048 * 2);
  bf16_t* w2t = (bf16_t*)alloc(2048 * 256 * 2);

  // weight conversions (pre-transposed to [N][K] bf16)
  cvt_transpose_bf16_kernel<<<(65536 + 255) / 256, 256, 0, stream>>>(wq, wqt, 256, 256);
  cvt_transpose_bf16_kernel<<<(65536 + 255) / 256, 256, 0, stream>>>(wk, wkt, 256, 256);
  cvt_transpose_bf16_kernel<<<(65536 + 255) / 256, 256, 0, stream>>>(wv, wvt, 256, 256);
  cvt_transpose_bf16_kernel<<<(65536 + 255) / 256, 256, 0, stream>>>(wm, wmt, 256, 256);
  cvt_transpose_bf16_kernel<<<(1048576 + 255) / 256, 256, 0, stream>>>(w1, w1t, 512, 2048);
  cvt_transpose_bf16_kernel<<<(524288 + 255) / 256, 256, 0, stream>>>(w2, w2t, 2048, 256);
  int nact = (int)(R * 256);
  cvt_f32_bf16_strided_kernel<<<(nact + 255) / 256, 256, 0, stream>>>(
      src, concat, 256, 512, nact);
  cvt_f32_bf16_kernel<<<(nact + 255) / 256, 256, 0, stream>>>(tgt, tgtb, nact);

  // QKV projections; epilogue applies roll+window permutation (q also *1/sqrt(C));
  // V additionally stored transposed per window.
  dim3 gq(256 / 128, RTOT / 128);
  gemm_bf16_kernel<EPI_QKV><<<gq, 256, 0, stream>>>(concat, 512, wqt, qwb, 256,
                                                    256, 0.0625f);
  gemm_bf16_kernel<EPI_QKV><<<gq, 256, 0, stream>>>(tgtb, 256, wkt, kwb, 256,
                                                    256, 1.0f);
  gemm_bf16_kernel<EPI_VT><<<gq, 256, 0, stream>>>(tgtb, 256, wvt, vwt, 256,
                                                   256, 1.0f);

  // fused masked window attention (flash), writes un-windowed bf16
  window_attention_kernel<<<dim3(128, 8), 256, 0, stream>>>(qwb, kwb, vwt, mask,
                                                            amsg);

  // merge projection -> fp32, then LN1 -> concat second half (bf16)
  gemm_bf16_kernel<EPI_F32><<<gq, 256, 0, stream>>>(amsg, 256, wmt, mraw, 256,
                                                    256, 1.0f);
  layernorm_kernel<0><<<RTOT / 8, 256, 0, stream>>>(mraw, g1, b1, nullptr, concat);

  // MLP, chunked over 8 row blocks to bound scratch
  for (int ch = 0; ch < 8; ++ch) {
    const bf16_t* a1 = concat + (size_t)ch * 16384 * 512;
    gemm_bf16_kernel<EPI_GELU><<<dim3(2048 / 128, 16384 / 128), 256, 0, stream>>>(
        a1, 512, w1t, hid, 2048, 512, 1.0f);
    float* d2 = mraw + (size_t)ch * 16384 * 256;
    gemm_bf16_kernel<EPI_F32><<<dim3(256 / 128, 16384 / 128), 256, 0, stream>>>(
        hid, 2048, w2t, d2, 256, 2048, 1.0f);
  }

  // LN2 + residual -> fp32 output
  layernorm_kernel<1><<<RTOT / 8, 256, 0, stream>>>(mraw, g2, b2, src, out);
}
